// HierarchicalPolicy_73770358276673
// MI455X (gfx1250) — matrix operand order repaired and checked by reading, not compiled, not run on hardware
//
#include <hip/hip_runtime.h>

// ---------------- problem constants ----------------
#define PB 8192   // batch
#define PS 512    // state dim
#define PH 1024   // hidden
#define PE 8      // experts
#define PA 64     // actions (output = 2A = 128)
#define POUT2A 128

typedef __attribute__((ext_vector_type(16))) _Float16 v16h;
typedef __attribute__((ext_vector_type(8)))  float    v8f;

// ---------------- GEMM tiling ----------------
#define BM 128
#define BN 128
#define BK 32
#define LDS_STR 40   // halves per LDS row (32 data + 8 pad = 80 bytes)

// =====================================================================
// f32 -> f16 elementwise convert
// =====================================================================
__global__ void cvt16_kernel(const float* __restrict__ s, _Float16* __restrict__ d, int n) {
  for (int i = blockIdx.x * blockDim.x + threadIdx.x; i < n; i += gridDim.x * blockDim.x)
    d[i] = (_Float16)s[i];
}

// =====================================================================
// W[K][N] f32 -> Wt[N][K] f16 (per-slab for stacked expert weights)
// grid: (N/32, K/32, slabs), block: (32, 8)
// =====================================================================
__global__ void transpose_cvt_kernel(const float* __restrict__ W, _Float16* __restrict__ Wt,
                                     int K, int N) {
  __shared__ float tile[32][33];
  const size_t slab = (size_t)blockIdx.z * K * N;
  const float* Ws = W + slab;
  _Float16* Wts = Wt + slab;
  int n0 = blockIdx.x * 32, k0 = blockIdx.y * 32;
  for (int r = threadIdx.y; r < 32; r += 8)
    tile[r][threadIdx.x] = Ws[(size_t)(k0 + r) * N + n0 + threadIdx.x];
  __syncthreads();
  for (int r = threadIdx.y; r < 32; r += 8)
    Wts[(size_t)(n0 + r) * K + k0 + threadIdx.x] = (_Float16)tile[threadIdx.x][r];
}

// =====================================================================
// Tiled WMMA GEMM with async global->LDS double-buffered staging.
//   C[M,N] = act(A[M,K] @ Wt[N,K]^T + bias)
//  - offsets != null : grouped mode, expert e = blockIdx.z,
//    rows = [offsets[e], offsets[e+1]); weight slab e*N*K; bias slab e*N
//  - gather != null  : A row index = gather[pos] (first expert layer)
//  - out16 : store f16, else f32 ; relu : apply max(0,.)
// block = 256 threads (8 waves, wave32); grid = (maxMtiles, N/BN, E or 1)
// =====================================================================
__global__ __launch_bounds__(256) void gemm_wmma_kernel(
    const _Float16* __restrict__ A, const _Float16* __restrict__ Wt,
    const float* __restrict__ bias, void* __restrict__ Cout,
    int K, int N, int Mtotal,
    const int* __restrict__ offsets, const int* __restrict__ gather,
    int relu, int out16)
{
  __shared__ _Float16 smA[2][BM * LDS_STR];
  __shared__ _Float16 smB[2][BN * LDS_STR];

  const int t = threadIdx.x;
  const int e = blockIdx.z;

  int base = 0, cnt = Mtotal;
  if (offsets) { base = offsets[e]; cnt = offsets[e + 1] - base; }
  const int mt = blockIdx.x, nt = blockIdx.y;
  if (mt * BM >= cnt) return;   // uniform early-out

  const _Float16* W = Wt + (size_t)e * N * K;
  const float* bi = bias + (size_t)e * N;

  // ---- staging ownership: thread t handles rows (t>>2) and (t>>2)+64,
  //      16-byte segment (t&3) of the 64B row -> 4 async b128 per tile ----
  const int srow0 = t >> 2;
  const int srow1 = srow0 + 64;
  const int seg   = t & 3;

  int l0 = mt * BM + srow0; if (l0 >= cnt) l0 = 0;
  int l1 = mt * BM + srow1; if (l1 >= cnt) l1 = 0;
  int pos0 = base + l0, pos1 = base + l1;
  const int sr0 = gather ? gather[pos0] : pos0;
  const int sr1 = gather ? gather[pos1] : pos1;

  const _Float16* gA0 = A + (size_t)sr0 * K + seg * 8;
  const _Float16* gA1 = A + (size_t)sr1 * K + seg * 8;
  const _Float16* gW0 = W + (size_t)(nt * BN + srow0) * K + seg * 8;
  const _Float16* gW1 = W + (size_t)(nt * BN + srow1) * K + seg * 8;

  unsigned ldsA0[2], ldsA1[2], ldsB0[2], ldsB1[2];
  #pragma unroll
  for (int p = 0; p < 2; ++p) {
    ldsA0[p] = (unsigned)(uintptr_t)&smA[p][srow0 * LDS_STR + seg * 8];
    ldsA1[p] = (unsigned)(uintptr_t)&smA[p][srow1 * LDS_STR + seg * 8];
    ldsB0[p] = (unsigned)(uintptr_t)&smB[p][srow0 * LDS_STR + seg * 8];
    ldsB1[p] = (unsigned)(uintptr_t)&smB[p][srow1 * LDS_STR + seg * 8];
  }

  auto issue_tile = [&](int k0, int p) {
    asm volatile("global_load_async_to_lds_b128 %0, %1, off"
                 :: "v"(ldsA0[p]), "v"((unsigned long long)(uintptr_t)(gA0 + k0)) : "memory");
    asm volatile("global_load_async_to_lds_b128 %0, %1, off"
                 :: "v"(ldsA1[p]), "v"((unsigned long long)(uintptr_t)(gA1 + k0)) : "memory");
    asm volatile("global_load_async_to_lds_b128 %0, %1, off"
                 :: "v"(ldsB0[p]), "v"((unsigned long long)(uintptr_t)(gW0 + k0)) : "memory");
    asm volatile("global_load_async_to_lds_b128 %0, %1, off"
                 :: "v"(ldsB1[p]), "v"((unsigned long long)(uintptr_t)(gW1 + k0)) : "memory");
  };

  const int lane = t & 31;
  const int w    = t >> 5;
  const int wm   = w >> 2;     // 0..1
  const int wn   = w & 3;      // 0..3
  const int hs   = lane >> 4;  // half-select
  const int lr   = lane & 15;

  v8f acc[4][2];
  #pragma unroll
  for (int f = 0; f < 4; ++f)
    #pragma unroll
    for (int g = 0; g < 2; ++g) {
      v8f z = {};
      acc[f][g] = z;
    }

  issue_tile(0, 0);                       // prologue: tile 0 in flight
  const int nk = K / BK;

  for (int ki = 0; ki < nk; ++ki) {
    const int p = ki & 1;
    if (ki + 1 < nk) {
      issue_tile((ki + 1) * BK, p ^ 1);   // next tile in flight (4 newest)
      asm volatile("s_wait_asynccnt 0x4" ::: "memory");  // this tile landed
    } else {
      asm volatile("s_wait_asynccnt 0x0" ::: "memory");
    }
    __syncthreads();                      // cross-wave visibility of LDS tile

    // ---- load fragments (CDNA5 16-bit WMMA layouts) ----
    v16h af[4], bf[2];
    #pragma unroll
    for (int f = 0; f < 4; ++f) {
      const _Float16* ap = &smA[p][(wm * 64 + f * 16 + lr) * LDS_STR];
      ((int4*)&af[f])[0] = *(const int4*)(ap + hs * 8);        // K = hs*8 .. +7
      ((int4*)&af[f])[1] = *(const int4*)(ap + 16 + hs * 8);   // K = 16+hs*8 ..
    }
    #pragma unroll
    for (int g = 0; g < 2; ++g) {
      const _Float16* bp = &smB[p][(wn * 32 + g * 16 + lr) * LDS_STR];
      ((int4*)&bf[g])[0] = *(const int4*)(bp + hs * 8);        // K = hs*8 .. +7
      ((int4*)&bf[g])[1] = *(const int4*)(bp + 16 + hs * 8);   // K = 16+hs*8 ..
    }

    #pragma unroll
    for (int f = 0; f < 4; ++f)
      #pragma unroll
      for (int g = 0; g < 2; ++g)
        acc[f][g] = __builtin_amdgcn_wmma_f32_16x16x32_f16(
            false, af[f], false, bf[g], (short)0, acc[f][g], false, false);

    __syncthreads();   // all waves done reading buf p before it is refilled
  }

  // ---- epilogue: bias + relu + store (C/D: vgpr r -> M = hs*8+r, lane lr -> N) ----
  #pragma unroll
  for (int f = 0; f < 4; ++f) {
    int mloc = wm * 64 + f * 16 + hs * 8;
    #pragma unroll
    for (int g = 0; g < 2; ++g) {
      int ncol = nt * BN + wn * 32 + g * 16 + lr;
      float bv = bi[ncol];
      #pragma unroll
      for (int r = 0; r < 8; ++r) {
        int m = mloc + r;
        int rrow = mt * BM + m;
        if (rrow < cnt) {
          float v = acc[f][g][r] + bv;
          if (relu) v = fmaxf(v, 0.f);
          size_t o = (size_t)(base + rrow) * N + ncol;
          if (out16) ((_Float16*)Cout)[o] = (_Float16)v;
          else       ((float*)Cout)[o] = v;
        }
      }
    }
  }
}

// =====================================================================
// JAX threefry2x32 (key = {0, seed}) — exact bit replication
// =====================================================================
__device__ inline uint2 threefry2x32(unsigned k0, unsigned k1, unsigned x0, unsigned x1) {
  unsigned ks2 = k0 ^ k1 ^ 0x1BD11BDAu;
  unsigned ks[3] = {k0, k1, ks2};
  x0 += k0; x1 += k1;
  const int R0[4] = {13, 15, 26, 6};
  const int R1[4] = {17, 29, 16, 24};
  #pragma unroll
  for (int g = 0; g < 5; ++g) {
    const int* R = (g & 1) ? R1 : R0;
    #pragma unroll
    for (int r = 0; r < 4; ++r) {
      x0 += x1;
      x1 = (x1 << R[r]) | (x1 >> (32 - R[r]));
      x1 ^= x0;
    }
    x0 += ks[(g + 1) % 3];
    x1 += ks[(g + 2) % 3] + (unsigned)(g + 1);
  }
  uint2 out; out.x = x0; out.y = x1; return out;
}

__device__ inline float bits_to_uniform(unsigned bits) {
  unsigned u = (bits >> 9) | 0x3f800000u;
  float f = __uint_as_float(u) - 1.0f;
  const float tiny = 1.17549435e-38f;
  f = f * (1.0f - tiny) + tiny;
  return fmaxf(f, tiny);
}

// =====================================================================
// Option head: logits = H2 @ Wo3 + bo3 ; softmax -> probs ; gumbel argmax -> sel
// one wave (32 lanes) per batch row; block 256 -> 8 rows; also counts experts
// =====================================================================
__global__ __launch_bounds__(256) void option_head_kernel(
    const _Float16* __restrict__ H2, const float* __restrict__ Wo3,
    const float* __restrict__ bo3,
    float* __restrict__ probs_out, float* __restrict__ selF_out,
    int* __restrict__ sel_out, int* __restrict__ counts)
{
  const int w = threadIdx.x >> 5, lane = threadIdx.x & 31;
  const int b = blockIdx.x * 8 + w;
  if (b >= PB) return;

  float acc[PE];
  #pragma unroll
  for (int e = 0; e < PE; ++e) acc[e] = 0.f;

  for (int k = lane; k < PH; k += 32) {
    float h = (float)H2[(size_t)b * PH + k];
    #pragma unroll
    for (int e = 0; e < PE; ++e) acc[e] += h * Wo3[k * PE + e];
  }
  #pragma unroll
  for (int off = 16; off; off >>= 1)
    #pragma unroll
    for (int e = 0; e < PE; ++e) acc[e] += __shfl_xor(acc[e], off, 32);

  if (lane == 0) {
    float l[PE];
    #pragma unroll
    for (int e = 0; e < PE; ++e) l[e] = acc[e] + bo3[e];
    float mx = l[0];
    #pragma unroll
    for (int e = 1; e < PE; ++e) mx = fmaxf(mx, l[e]);
    float s = 0.f, ex[PE];
    #pragma unroll
    for (int e = 0; e < PE; ++e) { ex[e] = expf(l[e] - mx); s += ex[e]; }
    float inv = 1.0f / s;
    #pragma unroll
    for (int e = 0; e < PE; ++e) probs_out[(size_t)b * PE + e] = ex[e] * inv;

    // jax.random.categorical(key(42), logits): argmax(logits + gumbel)
    const unsigned half = (PB * PE) / 2;  // 32768
    int best = 0; float bz = -3.4e38f;
    #pragma unroll
    for (int e = 0; e < PE; ++e) {
      unsigned i = (unsigned)(b * PE + e);
      unsigned bits;
      if (i < half) bits = threefry2x32(0u, 42u, i, i + half).x;
      else          bits = threefry2x32(0u, 42u, i - half, i).y;
      float u = bits_to_uniform(bits);
      float z = l[e] - logf(-logf(u));
      if (z > bz) { bz = z; best = e; }
    }
    sel_out[b] = best;
    selF_out[b] = (float)best;
    atomicAdd(&counts[best], 1);
  }
}

// =====================================================================
// Binning helpers
// =====================================================================
__global__ void zero_counts_kernel(int* counts) {
  if (threadIdx.x < PE) counts[threadIdx.x] = 0;
}

__global__ void scan_kernel(const int* __restrict__ counts, int* __restrict__ offsets,
                            int* __restrict__ cursor) {
  if (threadIdx.x == 0) {
    int o = 0;
    for (int e = 0; e < PE; ++e) { offsets[e] = o; cursor[e] = o; o += counts[e]; }
    offsets[PE] = o;
  }
}

__global__ void scatter_kernel(const int* __restrict__ sel, int* __restrict__ cursor,
                               int* __restrict__ perm) {
  int b = blockIdx.x * blockDim.x + threadIdx.x;
  if (b < PB) {
    int slot = atomicAdd(&cursor[sel[b]], 1);
    perm[slot] = b;
  }
}

// =====================================================================
// Finalize: un-permute expert outputs; split mean / exp(log_std)
// =====================================================================
__global__ void finalize_kernel(const float* __restrict__ Outc, const int* __restrict__ perm,
                                float* __restrict__ mean_out, float* __restrict__ std_out) {
  int idx = blockIdx.x * blockDim.x + threadIdx.x;
  if (idx >= PB * PA) return;
  int pos = idx >> 6, j = idx & 63;
  int b = perm[pos];
  mean_out[(size_t)b * PA + j] = Outc[(size_t)pos * POUT2A + j];
  std_out[(size_t)b * PA + j]  = expf(Outc[(size_t)pos * POUT2A + PA + j]);
}

// =====================================================================
// Host launcher
// =====================================================================
extern "C" void kernel_launch(void* const* d_in, const int* in_sizes, int n_in,
                              void* d_out, int out_size, void* d_ws, size_t ws_size,
                              hipStream_t stream) {
  (void)in_sizes; (void)n_in; (void)out_size; (void)ws_size;

  const float* state = (const float*)d_in[0];
  const float* Wo1 = (const float*)d_in[1];
  const float* bo1 = (const float*)d_in[2];
  const float* Wo2 = (const float*)d_in[3];
  const float* bo2 = (const float*)d_in[4];
  const float* Wo3 = (const float*)d_in[5];
  const float* bo3 = (const float*)d_in[6];
  const float* We1 = (const float*)d_in[7];
  const float* be1 = (const float*)d_in[8];
  const float* We2 = (const float*)d_in[9];
  const float* be2 = (const float*)d_in[10];
  const float* We3 = (const float*)d_in[11];
  const float* be3 = (const float*)d_in[12];

  // ---- workspace carve-up ----
  char* ws = (char*)d_ws;
  size_t off = 0;
  auto alloc = [&](size_t bytes) -> char* {
    char* p = ws + off;
    off = (off + bytes + 255) & ~(size_t)255;
    return p;
  };
  _Float16* Xh   = (_Float16*)alloc((size_t)PB * PS * 2);
  _Float16* Wo1t = (_Float16*)alloc((size_t)PS * PH * 2);
  _Float16* Wo2t = (_Float16*)alloc((size_t)PH * PH * 2);
  _Float16* We1t = (_Float16*)alloc((size_t)PE * PS * PH * 2);
  _Float16* We2t = (_Float16*)alloc((size_t)PE * PH * PH * 2);
  _Float16* We3t = (_Float16*)alloc((size_t)PE * PH * POUT2A * 2);
  _Float16* H1o  = (_Float16*)alloc((size_t)PB * PH * 2);
  _Float16* H2o  = (_Float16*)alloc((size_t)PB * PH * 2);
  _Float16* H1c  = (_Float16*)alloc((size_t)PB * PH * 2);
  _Float16* H2c  = (_Float16*)alloc((size_t)PB * PH * 2);
  float*    Outc = (float*)alloc((size_t)PB * POUT2A * 4);
  int*      sel     = (int*)alloc((size_t)PB * 4);
  int*      perm    = (int*)alloc((size_t)PB * 4);
  int*      counts  = (int*)alloc(PE * 4);
  int*      offsets = (int*)alloc((PE + 1) * 4);
  int*      cursor  = (int*)alloc(PE * 4);

  float* out_probs = (float*)d_out;                // [B,E]
  float* out_selF  = out_probs + (size_t)PB * PE;  // [B]
  float* out_mean  = out_selF + PB;                // [B,A]
  float* out_std   = out_mean + (size_t)PB * PA;   // [B,A]

  // ---- 1) precision conversion + weight transposes ----
  cvt16_kernel<<<2048, 256, 0, stream>>>(state, Xh, PB * PS);
  transpose_cvt_kernel<<<dim3(PH / 32, PS / 32, 1), dim3(32, 8), 0, stream>>>(Wo1, Wo1t, PS, PH);
  transpose_cvt_kernel<<<dim3(PH / 32, PH / 32, 1), dim3(32, 8), 0, stream>>>(Wo2, Wo2t, PH, PH);
  transpose_cvt_kernel<<<dim3(PH / 32, PS / 32, PE), dim3(32, 8), 0, stream>>>(We1, We1t, PS, PH);
  transpose_cvt_kernel<<<dim3(PH / 32, PH / 32, PE), dim3(32, 8), 0, stream>>>(We2, We2t, PH, PH);
  transpose_cvt_kernel<<<dim3(POUT2A / 32, PH / 32, PE), dim3(32, 8), 0, stream>>>(We3, We3t, PH, POUT2A);

  const int MT = PB / BM;  // 64 m-tiles

  // ---- 2) option MLP (layers 1-2 via WMMA, head fused with sampling) ----
  gemm_wmma_kernel<<<dim3(MT, PH / BN, 1), 256, 0, stream>>>(
      Xh, Wo1t, bo1, H1o, PS, PH, PB, nullptr, nullptr, 1, 1);
  gemm_wmma_kernel<<<dim3(MT, PH / BN, 1), 256, 0, stream>>>(
      H1o, Wo2t, bo2, H2o, PH, PH, PB, nullptr, nullptr, 1, 1);

  zero_counts_kernel<<<1, 32, 0, stream>>>(counts);
  option_head_kernel<<<PB / 8, 256, 0, stream>>>(H2o, Wo3, bo3, out_probs, out_selF, sel, counts);

  // ---- 3) bin rows by sampled expert ----
  scan_kernel<<<1, 32, 0, stream>>>(counts, offsets, cursor);
  scatter_kernel<<<PB / 256, 256, 0, stream>>>(sel, cursor, perm);

  // ---- 4) grouped expert GEMMs (only the selected expert per row: 8x fewer FLOPs) ----
  gemm_wmma_kernel<<<dim3(MT, PH / BN, PE), 256, 0, stream>>>(
      Xh, We1t, be1, H1c, PS, PH, PB, offsets, perm, 1, 1);
  gemm_wmma_kernel<<<dim3(MT, PH / BN, PE), 256, 0, stream>>>(
      H1c, We2t, be2, H2c, PH, PH, PB, offsets, nullptr, 1, 1);
  gemm_wmma_kernel<<<dim3(MT, POUT2A / BN, PE), 256, 0, stream>>>(
      H2c, We3t, be3, Outc, PH, POUT2A, PB, offsets, nullptr, 0, 0);

  // ---- 5) gather back + std = exp(log_std) ----
  finalize_kernel<<<(PB * PA + 255) / 256, 256, 0, stream>>>(Outc, perm, out_mean, out_std);
}
